// SparseTriangleMultiplicativeUpdate_20392504721478
// MI455X (gfx1250) — matrix operand order repaired and checked by reading, not compiled, not run on hardware
//
#include <hip/hip_runtime.h>

#define N_NODES 1024
#define DEG     24
#define K_NN    12
#define C_S     384
#define C_Z     128
#define C_G     16
#define N_RBF   64
#define N_EDGES (N_NODES * DEG)
#define N_EE    (N_EDGES * K_NN)

typedef __attribute__((ext_vector_type(16))) _Float16 v16h;
typedef __attribute__((ext_vector_type(8)))  float    v8f;

union FragH { v16h v; _Float16 h[16]; unsigned u[8]; };

__device__ __forceinline__ float sigmoidf_(float x) {
    return 1.0f / (1.0f + __expf(-x));
}

__device__ __forceinline__ v8f wmma_f16(v16h a, v16h b, v8f c) {
    // D = A(16x32 f16) * B(32x16 f16) + C(16x16 f32)
    return __builtin_amdgcn_wmma_f32_16x16x32_f16(
        /*neg_a=*/false, a, /*neg_b=*/false, b,
        /*c_mod=*/(short)0, c, /*reuse_a=*/false, /*reuse_b=*/false);
}

// ---- A fragment (16x32 f16) from row-major halves in LDS. ----
// Layout (ISA 7.12.2, 16-bit A 16x32): lanes 0-15 -> M=lane, K pairs {0..7,16..23};
// lanes 16-31 -> M=lane-16, K pairs {8..15,24..31}. 2 halves per VGPR slot.
__device__ __forceinline__ v16h frag_a_lds(const _Float16* tile, int stride) {
    const int lane = threadIdx.x & 31;
    const int m    = lane & 15;
    const int kb   = (lane & 16) ? 8 : 0;
    const unsigned* row32 = reinterpret_cast<const unsigned*>(tile + m * stride + kb);
    FragH f;
#pragma unroll
    for (int j = 0; j < 4; ++j) {
        f.u[j]     = row32[j];       // K = kb+2j, kb+2j+1
        f.u[4 + j] = row32[8 + j];   // K = 16+kb+2j, +1
    }
    return f.v;
}

// ---- B fragment (32x16 f16) from row-major Bt[N][K] halves in LDS. ----
// Layout: lane = N (mod 16); lanes 0-15 hold K {0..7,16..23},
// lanes 16-31 hold K {8..15,24..31}, pairs per VGPR (same as A).
__device__ __forceinline__ v16h frag_b_lds(const _Float16* tileT, int stride) {
    const int lane = threadIdx.x & 31;
    const unsigned* row32 = reinterpret_cast<const unsigned*>(
        tileT + (lane & 15) * stride + ((lane & 16) ? 8 : 0));
    FragH f;
#pragma unroll
    for (int j = 0; j < 4; ++j) { f.u[j] = row32[j]; f.u[4 + j] = row32[8 + j]; }
    return f.v;
}

// ---- B fragment directly from a global f32 row-major W[K][ldw] matrix. ----
__device__ __forceinline__ v16h frag_b_global(const float* __restrict__ W, int ldw,
                                              int n0, int k0) {
    const int lane = threadIdx.x & 31;
    const int n  = n0 + (lane & 15);
    const int kb = k0 + ((lane & 16) ? 8 : 0);
    FragH f;
#pragma unroll
    for (int j = 0; j < 8; ++j) {
        f.h[2 * j]     = (_Float16)W[(kb + 2 * j + (j >= 4 ? 8 : 0)) * ldw + n];
        f.h[2 * j + 1] = (_Float16)W[(kb + 2 * j + 1 + (j >= 4 ? 8 : 0)) * ldw + n];
    }
    return f.v;
}

__device__ __forceinline__ float rbf_(float d, int c) {
    const float mu = (20.0f / 63.0f) * (float)c;   // linspace(0,20,64)
    const float z  = (d - mu) * (64.0f / 20.0f);   // sigma = 20/64
    return __expf(-z * z);
}

// ================= Kernel 1: nl/nr node projections (tiny) =================
__global__ __launch_bounds__(256) void k1_node_proj(
    const float* __restrict__ nf,
    const float* __restrict__ wl_w, const float* __restrict__ wl_b,
    const float* __restrict__ wr_w, const float* __restrict__ wr_b,
    float* __restrict__ nl, float* __restrict__ nr) {
    const int idx   = blockIdx.x * 256 + threadIdx.x;   // 32768 total
    const int which = idx >> 14;                        // 0 = nl, 1 = nr
    const int node  = (idx & 16383) >> 4;
    const int o     = idx & 15;
    const float* W = which ? wr_w : wl_w;
    const float* B = which ? wr_b : wl_b;
    float s = B[o];
    const float* x = nf + (size_t)node * C_S;
#pragma unroll 4
    for (int k = 0; k < C_S; ++k) s += x[k] * W[k * C_G + o];
    (which ? nr : nl)[node * C_G + o] = s;
}

// ====== Kernel 2: ef = LN(edge_features); e2 & og gates (3 WMMA GEMMs) ======
__global__ __launch_bounds__(256) void k2_edge_gates(
    const float* __restrict__ ef_in,
    const float* __restrict__ ln_w, const float* __restrict__ ln_b,
    const float* __restrict__ ep_w, const float* __restrict__ ep_b,
    const float* __restrict__ eg_w, const float* __restrict__ eg_b,
    const float* __restrict__ og_w, const float* __restrict__ og_b,
    float* __restrict__ e2_out, float* __restrict__ og_out) {
    __shared__ _Float16 wT[128 * 136];    // transposed f16 weights Wt[n][k]
    __shared__ _Float16 efh[64 * 136];    // layernormed edges, f16
    const int tid  = threadIdx.x;
    const int wave = tid >> 5, lane = tid & 31;
    const int e0   = blockIdx.x * 64;

    // Layernorm: each wave handles 8 rows, lane-shuffle reduction (wave32).
    for (int rr = 0; rr < 8; ++rr) {
        const int row = wave * 8 + rr;
        const float* x = ef_in + (size_t)(e0 + row) * C_Z;
        float v0 = x[lane], v1 = x[lane + 32], v2 = x[lane + 64], v3 = x[lane + 96];
        float s  = v0 + v1 + v2 + v3;
        float ss = v0 * v0 + v1 * v1 + v2 * v2 + v3 * v3;
#pragma unroll
        for (int o = 16; o; o >>= 1) {
            s  += __shfl_xor(s, o, 32);
            ss += __shfl_xor(ss, o, 32);
        }
        const float mean = s * (1.0f / 128.0f);
        const float rstd = rsqrtf(ss * (1.0f / 128.0f) - mean * mean + 1e-5f);
        efh[row * 136 + lane]      = (_Float16)((v0 - mean) * rstd * ln_w[lane]      + ln_b[lane]);
        efh[row * 136 + lane + 32] = (_Float16)((v1 - mean) * rstd * ln_w[lane + 32] + ln_b[lane + 32]);
        efh[row * 136 + lane + 64] = (_Float16)((v2 - mean) * rstd * ln_w[lane + 64] + ln_b[lane + 64]);
        efh[row * 136 + lane + 96] = (_Float16)((v3 - mean) * rstd * ln_w[lane + 96] + ln_b[lane + 96]);
    }

    auto stage = [&](const float* __restrict__ W) {
        __syncthreads();
        for (int i = tid; i < 128 * 128; i += 256) {
            const int k = i >> 7, n = i & 127;
            wT[n * 136 + k] = (_Float16)W[i];
        }
        __syncthreads();
    };

    const int n0  = wave * 16;              // 8 waves -> 8 N-tiles
    const int col = n0 + (lane & 15);
    auto gemm_tile = [&](int mt) -> v8f {
        v8f acc = {};
#pragma unroll
        for (int kk = 0; kk < 4; ++kk)      // K = 128 -> 4 steps of 32
            acc = wmma_f16(frag_a_lds(efh + mt * 16 * 136 + kk * 32, 136),
                           frag_b_lds(wT + n0 * 136 + kk * 32, 136), acc);
        return acc;
    };

    v8f G[4];
    stage(eg_w);
    for (int mt = 0; mt < 4; ++mt) G[mt] = gemm_tile(mt);

    stage(ep_w);
    for (int mt = 0; mt < 4; ++mt) {
        v8f P = gemm_tile(mt);
#pragma unroll
        for (int r = 0; r < 8; ++r) {
            const int row = e0 + mt * 16 + r + ((lane & 16) ? 8 : 0);
            e2_out[(size_t)row * C_Z + col] =
                (P[r] + ep_b[col]) * sigmoidf_(G[mt][r] + eg_b[col]);
        }
    }

    stage(og_w);
    for (int mt = 0; mt < 4; ++mt) {
        v8f O = gemm_tile(mt);
#pragma unroll
        for (int r = 0; r < 8; ++r) {
            const int row = e0 + mt * 16 + r + ((lane & 16) ? 8 : 0);
            og_out[(size_t)row * C_Z + col] = sigmoidf_(O[r] + og_b[col]);
        }
    }
}

// == Kernel 3: fused outer-product GEMM + RBF GEMM + gating + segment-sum ==
// One block per destination node (24 edges, 288 pairs). Wave w owns N-tile w.
__global__ __launch_bounds__(256) void k3_pairs(
    const float* __restrict__ nl, const float* __restrict__ nr,
    const float* __restrict__ trans,
    const float* __restrict__ dg_w, const float* __restrict__ dg_b,
    const float* __restrict__ dp_w, const float* __restrict__ dp_b,
    const int* __restrict__ edge_index,   // row 0 = src
    const int* __restrict__ eei,          // [2, N_EE] flat
    const float* __restrict__ e2g, float* __restrict__ upd_raw) {
    __shared__ float e2s[DEG * C_Z];
    __shared__ float accs[DEG * C_Z];
    const int g    = blockIdx.x;
    const int tid  = threadIdx.x;
    const int wave = tid >> 5, lane = tid & 31;

    for (int i = tid; i < DEG * C_Z; i += 256) {
        e2s[i]  = e2g[(size_t)g * DEG * C_Z + i];
        accs[i] = 0.0f;
    }

    const int n0  = wave * 16;
    const int col = n0 + (lane & 15);
    // Weight B-fragments: loaded once, reused for all 18 M-tiles of the group.
    v16h Bdg[8], Bdp[2];
#pragma unroll
    for (int kk = 0; kk < 8; ++kk) Bdg[kk] = frag_b_global(dg_w, C_Z, n0, kk * 32);
    Bdp[0] = frag_b_global(dp_w, C_Z, n0, 0);
    Bdp[1] = frag_b_global(dp_w, C_Z, n0, 32);
    const float dgb = dg_b[col], dpb = dp_b[col];
    __syncthreads();

    for (int t = 0; t < (DEG * K_NN) / 16; ++t) {       // 18 M-tiles of 16 pairs
        const int p   = g * (DEG * K_NN) + t * 16 + (lane & 15);
        const int ee0 = eei[p];
        const int ee1 = eei[N_EE + p];
        const int s1  = edge_index[ee1];                // src of target edge
        const int s2  = edge_index[ee0];                // src of source edge
        int a0 = ee0 - g * DEG;                         // local edge in group
        a0 = a0 < 0 ? 0 : (a0 > DEG - 1 ? DEG - 1 : a0);

        float nlv[16];
        const float4* np4 = (const float4*)(nl + (size_t)s1 * C_G);
#pragma unroll
        for (int q = 0; q < 4; ++q) {
            float4 f = np4[q];
            nlv[4 * q] = f.x; nlv[4 * q + 1] = f.y; nlv[4 * q + 2] = f.z; nlv[4 * q + 3] = f.w;
        }
        const int kb = (lane & 16) ? 8 : 0;
        float nr8[8];
        const float4* rp4 = (const float4*)(nr + (size_t)s2 * C_G + kb);
        { float4 f = rp4[0]; nr8[0] = f.x; nr8[1] = f.y; nr8[2] = f.z; nr8[3] = f.w;
          f = rp4[1];        nr8[4] = f.x; nr8[5] = f.y; nr8[6] = f.z; nr8[7] = f.w; }

        const float dx = trans[s1 * 3]     - trans[s2 * 3]     + 1e-8f;
        const float dy = trans[s1 * 3 + 1] - trans[s2 * 3 + 1] + 1e-8f;
        const float dz = trans[s1 * 3 + 2] - trans[s2 * 3 + 2] + 1e-8f;
        const float d  = sqrtf(dx * dx + dy * dy + dz * dz);

        // gate3 = e3 @ dg_w : A built in-registers from the outer product.
        v8f accd = {};
#pragma unroll
        for (int kk = 0; kk < 8; ++kk) {                // K = 256 -> 8 steps
            FragH A;
            const float lo = nlv[2 * kk], hi = nlv[2 * kk + 1];
#pragma unroll
            for (int j = 0; j < 4; ++j) {
                A.h[2 * j]         = (_Float16)(lo * nr8[2 * j]);
                A.h[2 * j + 1]     = (_Float16)(lo * nr8[2 * j + 1]);
                A.h[8 + 2 * j]     = (_Float16)(hi * nr8[2 * j]);
                A.h[8 + 2 * j + 1] = (_Float16)(hi * nr8[2 * j + 1]);
            }
            accd = wmma_f16(A.v, Bdg[kk], accd);
        }

        // proj = rbf @ dp_w : A built from v_exp on the fly.
        v8f accr = {};
#pragma unroll
        for (int kk = 0; kk < 2; ++kk) {                // K = 64 -> 2 steps
            FragH A;
#pragma unroll
            for (int j = 0; j < 4; ++j) {
                const int c0 = kk * 32 + kb + 2 * j;
                const int c1 = kk * 32 + 16 + kb + 2 * j;
                A.h[2 * j]         = (_Float16)rbf_(d, c0);
                A.h[2 * j + 1]     = (_Float16)rbf_(d, c0 + 1);
                A.h[8 + 2 * j]     = (_Float16)rbf_(d, c1);
                A.h[8 + 2 * j + 1] = (_Float16)rbf_(d, c1 + 1);
            }
            accr = wmma_f16(A.v, Bdp[kk], accr);
        }

        // epilogue: sigmoid(gate3)*(proj+dp_b)*e2[eei0], segment-sum in LDS.
#pragma unroll
        for (int r = 0; r < 8; ++r) {
            const int mrow = r + ((lane & 16) ? 8 : 0);
            const int arow = __shfl(a0, mrow, 32);
            const float val = sigmoidf_(accd[r] + dgb) * (accr[r] + dpb) *
                              e2s[arow * C_Z + col];
            atomicAdd(&accs[arow * C_Z + col], val);    // ds_add_f32
        }
    }
    __syncthreads();
    for (int i = tid; i < DEG * C_Z; i += 256)
        upd_raw[(size_t)g * DEG * C_Z + i] = accs[i];
}

// ===== Kernel 4: LN(upd) @ lo_w + lo_b, gated by precomputed og-sigmoid =====
__global__ __launch_bounds__(256) void k4_out(
    const float* __restrict__ upd_raw,
    const float* __restrict__ lno_w, const float* __restrict__ lno_b,
    const float* __restrict__ lo_w, const float* __restrict__ lo_b,
    const float* __restrict__ og, float* __restrict__ out) {
    __shared__ _Float16 wT[128 * 136];
    __shared__ _Float16 uh[64 * 136];
    const int tid  = threadIdx.x;
    const int wave = tid >> 5, lane = tid & 31;
    const int e0   = blockIdx.x * 64;

    for (int rr = 0; rr < 8; ++rr) {
        const int row = wave * 8 + rr;
        const float* x = upd_raw + (size_t)(e0 + row) * C_Z;
        float v0 = x[lane], v1 = x[lane + 32], v2 = x[lane + 64], v3 = x[lane + 96];
        float s  = v0 + v1 + v2 + v3;
        float ss = v0 * v0 + v1 * v1 + v2 * v2 + v3 * v3;
#pragma unroll
        for (int o = 16; o; o >>= 1) {
            s  += __shfl_xor(s, o, 32);
            ss += __shfl_xor(ss, o, 32);
        }
        const float mean = s * (1.0f / 128.0f);
        const float rstd = rsqrtf(ss * (1.0f / 128.0f) - mean * mean + 1e-5f);
        uh[row * 136 + lane]      = (_Float16)((v0 - mean) * rstd * lno_w[lane]      + lno_b[lane]);
        uh[row * 136 + lane + 32] = (_Float16)((v1 - mean) * rstd * lno_w[lane + 32] + lno_b[lane + 32]);
        uh[row * 136 + lane + 64] = (_Float16)((v2 - mean) * rstd * lno_w[lane + 64] + lno_b[lane + 64]);
        uh[row * 136 + lane + 96] = (_Float16)((v3 - mean) * rstd * lno_w[lane + 96] + lno_b[lane + 96]);
    }
    __syncthreads();
    for (int i = tid; i < 128 * 128; i += 256) {
        const int k = i >> 7, n = i & 127;
        wT[n * 136 + k] = (_Float16)lo_w[i];
    }
    __syncthreads();

    const int n0 = wave * 16, col = n0 + (lane & 15);
    for (int mt = 0; mt < 4; ++mt) {
        v8f acc = {};
#pragma unroll
        for (int kk = 0; kk < 4; ++kk)
            acc = wmma_f16(frag_a_lds(uh + mt * 16 * 136 + kk * 32, 136),
                           frag_b_lds(wT + n0 * 136 + kk * 32, 136), acc);
#pragma unroll
        for (int r = 0; r < 8; ++r) {
            const int row = e0 + mt * 16 + r + ((lane & 16) ? 8 : 0);
            out[(size_t)row * C_Z + col] =
                (acc[r] + lo_b[col]) * og[(size_t)row * C_Z + col];
        }
    }
}

extern "C" void kernel_launch(void* const* d_in, const int* in_sizes, int n_in,
                              void* d_out, int out_size, void* d_ws, size_t ws_size,
                              hipStream_t stream) {
    const float* node_features = (const float*)d_in[0];
    const float* node_trans    = (const float*)d_in[1];
    const float* edge_features = (const float*)d_in[2];
    const float* ln_w  = (const float*)d_in[3];
    const float* ln_b  = (const float*)d_in[4];
    const float* wl_w  = (const float*)d_in[5];
    const float* wl_b  = (const float*)d_in[6];
    const float* wr_w  = (const float*)d_in[7];
    const float* wr_b  = (const float*)d_in[8];
    const float* ep_w  = (const float*)d_in[9];
    const float* ep_b  = (const float*)d_in[10];
    const float* eg_w  = (const float*)d_in[11];
    const float* eg_b  = (const float*)d_in[12];
    const float* dg_w  = (const float*)d_in[13];
    const float* dg_b  = (const float*)d_in[14];
    const float* dp_w  = (const float*)d_in[15];
    const float* dp_b  = (const float*)d_in[16];
    const float* lno_w = (const float*)d_in[17];
    const float* lno_b = (const float*)d_in[18];
    const float* lo_w  = (const float*)d_in[19];
    const float* lo_b  = (const float*)d_in[20];
    const float* og_w  = (const float*)d_in[21];
    const float* og_b  = (const float*)d_in[22];
    const int* edge_index = (const int*)d_in[23];
    const int* eei        = (const int*)d_in[24];

    size_t off = 0;
    char* wsb = (char*)d_ws;
    float* nl   = (float*)(wsb + off); off += (size_t)N_NODES * C_G * sizeof(float);
    float* nr   = (float*)(wsb + off); off += (size_t)N_NODES * C_G * sizeof(float);
    float* e2   = (float*)(wsb + off); off += (size_t)N_EDGES * C_Z * sizeof(float);
    float* ogg  = (float*)(wsb + off); off += (size_t)N_EDGES * C_Z * sizeof(float);
    float* updr = (float*)(wsb + off); off += (size_t)N_EDGES * C_Z * sizeof(float);
    (void)off; (void)ws_size; (void)in_sizes; (void)n_in; (void)out_size;

    k1_node_proj<<<(N_NODES * C_G * 2) / 256, 256, 0, stream>>>(
        node_features, wl_w, wl_b, wr_w, wr_b, nl, nr);
    k2_edge_gates<<<N_EDGES / 64, 256, 0, stream>>>(
        edge_features, ln_w, ln_b, ep_w, ep_b, eg_w, eg_b, og_w, og_b, e2, ogg);
    k3_pairs<<<N_NODES, 256, 0, stream>>>(
        nl, nr, node_trans, dg_w, dg_b, dp_w, dp_b, edge_index, eei, e2, updr);
    k4_out<<<N_EDGES / 64, 256, 0, stream>>>(
        updr, lno_w, lno_b, lo_w, lo_b, ogg, (float*)d_out);
}
